// MS_Attention_linear_56573309224414
// MI455X (gfx1250) — 16-bit and 8-bit matrix operand order repaired and checked by reading, compiled, not run on hardware
//
#include <hip/hip_runtime.h>
#include <hip/hip_fp16.h>

#define TT 4
#define BB 32
#define CC 256
#define NN 256
#define NH 8
#define HD 32
#define APITCH 272

typedef __attribute__((ext_vector_type(16))) _Float16 v16h;
typedef __attribute__((ext_vector_type(8)))  float    v8f;
typedef __attribute__((ext_vector_type(8)))  int      v8i;

#if __has_builtin(__builtin_amdgcn_global_load_async_to_lds_b128) && \
    __has_builtin(__builtin_amdgcn_s_wait_asynccnt)
#define HAVE_ASYNC_LDS 1
#endif

#ifdef HAVE_ASYNC_LDS
typedef int v4ig __attribute__((vector_size(16)));
__device__ static inline void async_b128(const void* g, void* l) {
  __builtin_amdgcn_global_load_async_to_lds_b128((v4ig*)(unsigned long long)g,
                                                 (v4ig*)l, 0, 0);
}
#endif

__global__ void lif_x_kernel(const float* __restrict__ x, _Float16* __restrict__ xs) {
  const int nper = BB * CC * NN;
  int i = blockIdx.x * blockDim.x + threadIdx.x;
  if (i >= nper) return;
  float v = 0.0f;
#pragma unroll
  for (int t = 0; t < TT; ++t) {
    float xt = x[(size_t)t * nper + i];
    v += (xt - v) * 0.5f;
    int s = (v >= 1.0f);
    xs[(size_t)t * nper + i] = (_Float16)(s ? 1.0f : 0.0f);
    v = s ? 0.0f : v;
  }
}

__global__ void lif_spike_u8_kernel(const _Float16* __restrict__ pre,
                                    unsigned char* __restrict__ spk,
                                    int token_major) {
  const int nper = BB * CC * NN;
  int i = blockIdx.x * blockDim.x + threadIdx.x;
  if (i >= nper) return;
  int n = i & (NN - 1);
  int c = (i >> 8) & (CC - 1);
  int b = i >> 16;
  int h = c >> 5, dd = c & 31;
  float v = 0.0f;
#pragma unroll
  for (int t = 0; t < TT; ++t) {
    float xt = (float)pre[(size_t)t * nper + i];
    v += (xt - v) * 0.5f;
    int s = (v >= 1.0f);
    size_t off;
    if (token_major)
      off = ((((size_t)t * BB + b) * NH + h) * NN + n) * HD + dd;
    else
      off = (size_t)t * nper + i;
    spk[off] = (unsigned char)s;
    v = s ? 0.0f : v;
  }
}

__global__ void lif_spike_f16_kernel(const _Float16* __restrict__ pre,
                                     _Float16* __restrict__ spk) {
  const int nper = BB * CC * NN;
  int i = blockIdx.x * blockDim.x + threadIdx.x;
  if (i >= nper) return;
  float v = 0.0f;
#pragma unroll
  for (int t = 0; t < TT; ++t) {
    float xt = (float)pre[(size_t)t * nper + i];
    v += (xt - v) * 0.5f;
    int s = (v >= 1.0f);
    spk[(size_t)t * nper + i] = (_Float16)(s ? 1.0f : 0.0f);
    v = s ? 0.0f : v;
  }
}

__global__ void conv_bn_wmma_kernel(const _Float16* __restrict__ X,
                                    const float* __restrict__ W,
                                    const float* __restrict__ gamma,
                                    const float* __restrict__ beta,
                                    const float* __restrict__ mean,
                                    const float* __restrict__ var,
                                    _Float16* __restrict__ out_h,
                                    float* __restrict__ out_f) {
  const int lane = threadIdx.x & 31;
  const int wv   = threadIdx.x >> 5;
  const int gwave = blockIdx.x * 8 + wv;
  const int tile = gwave & 255;
  const int tb   = gwave >> 8;
  const int d0 = (tile >> 4) * 16;
  const int n0 = (tile & 15) * 16;
  const int r = lane & 15;
  const int g = lane >> 4;

  const _Float16* Xb = X + (size_t)tb * CC * NN;
  v8f acc = {};

  const int drow = d0 + r;
  const int ncol = n0 + r;

#pragma unroll
  for (int k0 = 0; k0 < CC; k0 += 32) {
    v16h a, b;
    const float* wrow = W + (size_t)drow * CC + k0 + g * 16;
#pragma unroll
    for (int i = 0; i < 8; ++i) {
      a[i]     = (_Float16)wrow[i];
      a[8 + i] = (_Float16)wrow[8 + i];
    }
    const _Float16* xcol = Xb + (size_t)(k0 + g * 16) * NN + ncol;
#pragma unroll
    for (int i = 0; i < 16; ++i) b[i] = xcol[(size_t)i * NN];
    acc = __builtin_amdgcn_wmma_f32_16x16x32_f16(false, a, false, b,
                                                 (short)0, acc, false, false);
  }

#pragma unroll
  for (int rr = 0; rr < 8; ++rr) {
    int d = d0 + g * 8 + rr;
    float inv = gamma[d] / sqrtf(var[d] + 1e-5f);
    float val = (acc[rr] - mean[d]) * inv + beta[d];
    size_t off = ((size_t)tb * CC + d) * NN + (n0 + r);
    if (out_f) out_f[off] = val;
    else       out_h[off] = (_Float16)val;
  }
}

__global__ void attn_iu8_wmma_kernel(const unsigned char* __restrict__ qs,
                                     const unsigned char* __restrict__ ks,
                                     const unsigned char* __restrict__ vs,
                                     _Float16* __restrict__ outpre) {
  __shared__ __align__(16) unsigned char lds_k[NN * HD];
  __shared__ __align__(16) unsigned char lds_v[HD * NN];
  __shared__ __align__(16) unsigned char lds_q[64 * HD];
  __shared__ __align__(16) unsigned char lds_attn[64 * APITCH];

  const int tbh = blockIdx.x;
  const int h   = tbh & (NH - 1);
  const int tb  = tbh >> 3;
  const int lane = threadIdx.x & 31;
  const int wv   = threadIdx.x >> 5;
  const int r = lane & 15;
  const int g = lane >> 4;

  const unsigned char* qg = qs + (size_t)tbh * (NN * HD);
  const unsigned char* kg = ks + (size_t)tbh * (NN * HD);
  const unsigned char* vg = vs + ((size_t)tb * CC + h * HD) * NN;

#ifdef HAVE_ASYNC_LDS
  {
    unsigned tid16 = threadIdx.x * 16u;
    async_b128(kg + tid16,        lds_k + tid16);
    async_b128(kg + 4096 + tid16, lds_k + 4096 + tid16);
    async_b128(vg + tid16,        lds_v + tid16);
    async_b128(vg + 4096 + tid16, lds_v + 4096 + tid16);
    __builtin_amdgcn_s_wait_asynccnt(0);
  }
#else
  {
    const uint4* gk = (const uint4*)kg;
    const uint4* gv = (const uint4*)vg;
    uint4* sk = (uint4*)lds_k;
    uint4* sv = (uint4*)lds_v;
    sk[threadIdx.x]       = gk[threadIdx.x];
    sk[threadIdx.x + 256] = gk[threadIdx.x + 256];
    sv[threadIdx.x]       = gv[threadIdx.x];
    sv[threadIdx.x + 256] = gv[threadIdx.x + 256];
  }
#endif
  __syncthreads();

  const float sc = 2.0f * 0.17677669529663687f;

  for (int nb = 0; nb < 4; ++nb) {
#ifdef HAVE_ASYNC_LDS
    if (threadIdx.x < 128) {
      unsigned tid16 = threadIdx.x * 16u;
      async_b128(qg + nb * 2048 + tid16, lds_q + tid16);
    }
    __builtin_amdgcn_s_wait_asynccnt(0);
#else
    if (threadIdx.x < 128)
      ((uint4*)lds_q)[threadIdx.x] = ((const uint4*)(qg + nb * 2048))[threadIdx.x];
#endif
    __syncthreads();

    for (int tt = wv; tt < 64; tt += 8) {
      const int ti = tt >> 4;
      const int tj = tt & 15;
      v8i a = {}, b = {};
      const unsigned char* ap = lds_q + (ti * 16 + r) * HD + g * 8;
      a[0] = *(const int*)(ap + 0);
      a[1] = *(const int*)(ap + 4);
      a[2] = *(const int*)(ap + 16);
      a[3] = *(const int*)(ap + 20);
      const unsigned char* bp = lds_k + (tj * 16 + r) * HD + g * 8;
      b[0] = *(const int*)(bp + 0);
      b[1] = *(const int*)(bp + 4);
      b[2] = *(const int*)(bp + 16);
      b[3] = *(const int*)(bp + 20);
      v8i cacc = {};
      cacc = __builtin_amdgcn_wmma_i32_16x16x64_iu8(false, a, false, b, cacc, false, false);
#pragma unroll
      for (int rr = 0; rr < 8; ++rr)
        lds_attn[(ti * 16 + g * 8 + rr) * APITCH + tj * 16 + r] = (unsigned char)cacc[rr];
    }
    __syncthreads();

    {
      const int ri = wv >> 1;
      const int dj = wv & 1;
      v8i cacc = {};
#pragma unroll
      for (int kk = 0; kk < 4; ++kk) {
        v8i a, b;
        const unsigned char* ap = lds_attn + (ri * 16 + r) * APITCH + kk * 64 + g * 8;
        a[0] = *(const int*)(ap + 0);
        a[1] = *(const int*)(ap + 4);
        a[2] = *(const int*)(ap + 16);
        a[3] = *(const int*)(ap + 20);
        a[4] = *(const int*)(ap + 32);
        a[5] = *(const int*)(ap + 36);
        a[6] = *(const int*)(ap + 48);
        a[7] = *(const int*)(ap + 52);
        const unsigned char* bp = lds_v + (dj * 16 + r) * NN + kk * 64 + g * 8;
        b[0] = *(const int*)(bp + 0);
        b[1] = *(const int*)(bp + 4);
        b[2] = *(const int*)(bp + 16);
        b[3] = *(const int*)(bp + 20);
        b[4] = *(const int*)(bp + 32);
        b[5] = *(const int*)(bp + 36);
        b[6] = *(const int*)(bp + 48);
        b[7] = *(const int*)(bp + 52);
        cacc = __builtin_amdgcn_wmma_i32_16x16x64_iu8(false, a, false, b, cacc, false, false);
      }
#pragma unroll
      for (int rr = 0; rr < 8; ++rr) {
        int n  = nb * 64 + ri * 16 + g * 8 + rr;
        int dd = dj * 16 + r;
        size_t off = ((size_t)tb * CC + h * HD + dd) * NN + n;
        outpre[off] = (_Float16)((float)cacc[rr] * sc);
      }
    }
    __syncthreads();
  }
}

extern "C" void kernel_launch(void* const* d_in, const int* in_sizes, int n_in,
                              void* d_out, int out_size, void* d_ws, size_t ws_size,
                              hipStream_t stream) {
  (void)in_sizes; (void)n_in; (void)out_size; (void)ws_size;

  const float* x       = (const float*)d_in[0];
  const float* q_w     = (const float*)d_in[1];
  const float* q_gamma = (const float*)d_in[2];
  const float* q_beta  = (const float*)d_in[3];
  const float* q_mean  = (const float*)d_in[4];
  const float* q_var   = (const float*)d_in[5];
  const float* k_w     = (const float*)d_in[6];
  const float* k_gamma = (const float*)d_in[7];
  const float* k_beta  = (const float*)d_in[8];
  const float* k_mean  = (const float*)d_in[9];
  const float* k_var   = (const float*)d_in[10];
  const float* v_w     = (const float*)d_in[11];
  const float* v_gamma = (const float*)d_in[12];
  const float* v_beta  = (const float*)d_in[13];
  const float* v_mean  = (const float*)d_in[14];
  const float* v_var   = (const float*)d_in[15];
  const float* p_w     = (const float*)d_in[16];
  const float* p_gamma = (const float*)d_in[17];
  const float* p_beta  = (const float*)d_in[18];
  const float* p_mean  = (const float*)d_in[19];
  const float* p_var   = (const float*)d_in[20];

  const size_t SZ_F16 = (size_t)TT * BB * CC * NN * 2;
  const size_t SZ_U8  = (size_t)TT * BB * CC * NN;
  unsigned char* ws = (unsigned char*)d_ws;
  _Float16* xs      = (_Float16*)(ws);
  _Float16* preQ    = (_Float16*)(ws + 1 * SZ_F16);
  _Float16* preK    = (_Float16*)(ws + 2 * SZ_F16);
  _Float16* preV    = (_Float16*)(ws + 3 * SZ_F16);
  unsigned char* sq = ws + 4 * SZ_F16;
  unsigned char* sk = sq + SZ_U8;
  unsigned char* sv = sk + SZ_U8;
  _Float16* attnpre = (_Float16*)(sv + SZ_U8);
  _Float16* sattn   = (_Float16*)(sv + SZ_U8 + SZ_F16);

  const int nper   = BB * CC * NN;
  const int eblk   = 256;
  const int egrid  = nper / eblk;
  const int cgrid  = (TT * BB * 256) / 8;

  lif_x_kernel<<<egrid, eblk, 0, stream>>>(x, xs);

  conv_bn_wmma_kernel<<<cgrid, 256, 0, stream>>>(xs, q_w, q_gamma, q_beta, q_mean, q_var, preQ, nullptr);
  conv_bn_wmma_kernel<<<cgrid, 256, 0, stream>>>(xs, k_w, k_gamma, k_beta, k_mean, k_var, preK, nullptr);
  conv_bn_wmma_kernel<<<cgrid, 256, 0, stream>>>(xs, v_w, v_gamma, v_beta, v_mean, v_var, preV, nullptr);

  lif_spike_u8_kernel<<<egrid, eblk, 0, stream>>>(preQ, sq, 1);
  lif_spike_u8_kernel<<<egrid, eblk, 0, stream>>>(preK, sk, 1);
  lif_spike_u8_kernel<<<egrid, eblk, 0, stream>>>(preV, sv, 0);

  attn_iu8_wmma_kernel<<<TT * BB * NH, 256, 0, stream>>>(sq, sk, sv, attnpre);

  lif_spike_f16_kernel<<<egrid, eblk, 0, stream>>>(attnpre, sattn);

  conv_bn_wmma_kernel<<<cgrid, 256, 0, stream>>>(sattn, p_w, p_gamma, p_beta, p_mean, p_var,
                                                 nullptr, (float*)d_out);
}
